// SelfAttentionUnit_65481071404758
// MI455X (gfx1250) — matrix operand order repaired and checked by reading, not compiled, not run on hardware
//
#include <hip/hip_runtime.h>
#include <hip/hip_bf16.h>
#include <math.h>

// ---------------------------------------------------------------------------
// SelfAttentionUnit for MI455X (gfx1250, wave32, WMMA + TDM)
//   qkv   = x @ W_qkv                     (bf16 WMMA, f32 accum)
//   per-token 16x16 head-mix softmax      (VALU, one wave per token)
//   out   = attn @ W_out + b_out          (bf16 WMMA, f32 accum, f32 out)
// A-tiles staged into LDS by the Tensor Data Mover (tensor_load_to_lds),
// B-tiles staged transposed with pair-packed ds_store_b32.
// ---------------------------------------------------------------------------

typedef __attribute__((ext_vector_type(16))) __bf16 v16bf;
typedef __attribute__((ext_vector_type(8)))  __bf16 v8bf;
typedef __attribute__((ext_vector_type(8)))  float  v8f;
typedef __attribute__((ext_vector_type(8)))  unsigned short v8us;
typedef __attribute__((ext_vector_type(4)))  unsigned int v4u;
typedef __attribute__((ext_vector_type(8)))  int v8i;
typedef __attribute__((ext_vector_type(4)))  int v4i;

#define EMBED  2048
#define NHEADS 16
#define HDIM   128
#define BATCH  4
#define SEQ    4096
#define MROWS  (BATCH * SEQ)   // 16384
#define NQKV   (3 * EMBED)     // 6144

#define BM 128
#define BN 128
#define BK 32

#if __has_builtin(__builtin_amdgcn_tensor_load_to_lds) && \
    __has_builtin(__builtin_amdgcn_s_wait_tensorcnt)
#define HAVE_TDM 1
#else
#define HAVE_TDM 0
#endif

// ---------------- f32 -> bf16 convert ----------------
__global__ void f32_to_bf16_kernel(const float* __restrict__ in,
                                   __bf16* __restrict__ out, int n) {
  int i = blockIdx.x * blockDim.x + threadIdx.x;
  if (i < n) out[i] = (__bf16)in[i];
}

// ---------------- WMMA GEMM: C[M,N] = A[M,K] @ B[K,N] (+bias) ----------------
// A, B bf16 row-major. Block tile 128x128, 8 waves (2x4), wave tile 64x32.
template <bool OUT_BF16>
__global__ __launch_bounds__(256) void gemm_bf16_wmma(
    const __bf16* __restrict__ A, const __bf16* __restrict__ B,
    void* __restrict__ Cout, const float* __restrict__ bias,
    int M, int N, int K) {
  __shared__ __align__(16) __bf16 lA[BM * BK];  // (m, k) row-major
  __shared__ __align__(16) __bf16 lB[BN * BK];  // (n, k) K-transposed

  const int tid  = threadIdx.x;
  const int wave = tid >> 5;
  const int lane = tid & 31;
  const int bm = blockIdx.y * BM;
  const int bn = blockIdx.x * BN;
  const int wm = (wave >> 2) * 64;   // 0 or 64
  const int wn = (wave & 3) * 32;    // 0,32,64,96

  v8f acc[4][2] = {};

  // fragment addressing per ISA 7.12.2 (16-bit A 16x32 / B 32x16, wave32)
  const int ar = lane & 15;          // A row / B col within 16
  const int ak = (lane >> 4) * 8;    // A: lanes 0-15 -> K0..7/16..23; 16-31 -> +8
  const int bk = (lane >> 4) * 8;    // B: lanes 0-15 -> K0..7/16..23; 16-31 -> +8

  const unsigned KDW = (unsigned)(K >> 1);  // row length in dwords

  for (int k0 = 0; k0 < K; k0 += BK) {
#if HAVE_TDM
    // ---- stage A (128x32 bf16, row-major, row stride K*2 B) via TDM ----
    if (wave == 0) {
      unsigned ldsA = (unsigned)(size_t)(void*)&lA[0];  // low 32 bits = LDS offset
      unsigned long long ga =
          (unsigned long long)(size_t)(A + (size_t)bm * K + k0);
      v4u g0;
      g0[0] = 1u;                                     // count=1, no gather
      g0[1] = ldsA;                                   // lds_addr
      g0[2] = (unsigned)(ga & 0xFFFFFFFFu);           // global_addr[31:0]
      g0[3] = (unsigned)((ga >> 32) & 0x01FFFFFFu) |  // global_addr[56:32]
              (2u << 30);                             // type=2 ("image")
      v8i g1;
      g1[0] = (int)(2u << 16);                        // data_size=2 (4B units)
      g1[1] = (int)((KDW & 0xFFFFu) << 16);           // tensor_dim0[15:0]
      g1[2] = (int)(((KDW >> 16) & 0xFFFFu) |         // tensor_dim0[31:16]
                    (128u << 16));                    // tensor_dim1[15:0]=128
      g1[3] = (int)(16u << 16);                       // tile_dim0=16 dwords
      g1[4] = 128;                                    // tile_dim1=128, tile_dim2=0
      g1[5] = (int)KDW;                               // tensor_dim0_stride[31:0]
      g1[6] = 0;                                      // stride hi, dim1_stride lo
      g1[7] = 0;
      v4i z4 = {0, 0, 0, 0};
      v8i z8 = {0, 0, 0, 0, 0, 0, 0, 0};
      __builtin_amdgcn_tensor_load_to_lds(g0, g1, z4, z4, z8, 0);
      __builtin_amdgcn_s_wait_tensorcnt(0);
    }
#else
    // ---- fallback: stage A through VGPRs ----
    {
      int r = tid >> 1;
      int c = (tid & 1) * 16;
      const __bf16* src = A + (size_t)(bm + r) * K + k0 + c;
      *(v8bf*)&lA[r * BK + c]     = *(const v8bf*)(src);
      *(v8bf*)&lA[r * BK + c + 8] = *(const v8bf*)(src + 8);
    }
#endif
    // ---- stage B (32x128) -> LDS transposed (n,k), pair-packed b32 stores ----
    {
      int kp = tid >> 4;            // row pair 0..15 -> rows 2kp, 2kp+1
      int c0 = (tid & 15) * 8;      // 8 columns per thread
      const unsigned short* src =
          (const unsigned short*)B + (size_t)(k0 + 2 * kp) * N + bn + c0;
      v8us b0 = *(const v8us*)(src);
      v8us b1 = *(const v8us*)(src + N);
      unsigned* lBd = (unsigned*)lB;
#pragma unroll
      for (int j = 0; j < 8; ++j)
        lBd[(c0 + j) * (BK / 2) + kp] =
            (unsigned)b0[j] | ((unsigned)b1[j] << 16);
    }
    __syncthreads();

    v16bf af[4];
#pragma unroll
    for (int i = 0; i < 4; ++i) {
      int m = wm + i * 16 + ar;
      v8bf lo = *(const v8bf*)&lA[m * BK + ak];
      v8bf hi = *(const v8bf*)&lA[m * BK + ak + 16];
      af[i] = __builtin_shufflevector(lo, hi, 0, 1, 2, 3, 4, 5, 6, 7,
                                      8, 9, 10, 11, 12, 13, 14, 15);
    }
    v16bf bfr[2];
#pragma unroll
    for (int j = 0; j < 2; ++j) {
      int n = wn + j * 16 + ar;
      v8bf lo = *(const v8bf*)&lB[n * BK + bk];
      v8bf hi = *(const v8bf*)&lB[n * BK + bk + 16];
      bfr[j] = __builtin_shufflevector(lo, hi, 0, 1, 2, 3, 4, 5, 6, 7,
                                       8, 9, 10, 11, 12, 13, 14, 15);
    }

#pragma unroll
    for (int i = 0; i < 4; ++i)
#pragma unroll
      for (int j = 0; j < 2; ++j)
        acc[i][j] = __builtin_amdgcn_wmma_f32_16x16x32_bf16(
            false, af[i], false, bfr[j], (short)0, acc[i][j], false, false);

    __syncthreads();
  }

  // epilogue: C/D layout — VGPR r: lanes 0-15 M=r, lanes 16-31 M=r+8; N=lane&15
  const int mh = (lane >> 4) * 8;
  const int nc = lane & 15;
#pragma unroll
  for (int i = 0; i < 4; ++i) {
#pragma unroll
    for (int j = 0; j < 2; ++j) {
      int n = bn + wn + j * 16 + nc;
      float bv = bias ? bias[n] : 0.0f;
#pragma unroll
      for (int r = 0; r < 8; ++r) {
        int m = bm + wm + i * 16 + mh + r;
        float vv = acc[i][j][r] + bv;
        if (OUT_BF16)
          ((__bf16*)Cout)[(size_t)m * N + n] = (__bf16)vv;
        else
          ((float*)Cout)[(size_t)m * N + n] = vv;
      }
    }
  }
}

// ---------------- per-token head-mix softmax (one wave per token) ----------------
// scores[h][H] = (q[h,:] . k[H,:]) / sqrt(128); softmax over H; out = w @ v
__global__ __launch_bounds__(128) void attn_softmax_kernel(
    const __bf16* __restrict__ qkv, __bf16* __restrict__ attn) {
  __shared__ float sc[4][NHEADS][NHEADS];
  const int wave = threadIdx.x >> 5;
  const int lane = threadIdx.x & 31;
  const int pos  = blockIdx.x * 4 + wave;

  const __bf16* q = qkv + (size_t)pos * NQKV;
  const __bf16* k = q + EMBED;
  const __bf16* v = q + 2 * EMBED;

  const int h  = lane >> 1;          // each lane owns one q row, half the k rows
  const int H0 = (lane & 1) * 8;

  float s[8] = {0.f, 0.f, 0.f, 0.f, 0.f, 0.f, 0.f, 0.f};
  for (int c = 0; c < HDIM / 8; ++c) {
    v8bf qc = *(const v8bf*)(q + h * HDIM + c * 8);
    float qf[8];
#pragma unroll
    for (int t = 0; t < 8; ++t) qf[t] = (float)qc[t];
#pragma unroll
    for (int i = 0; i < 8; ++i) {
      v8bf kc = *(const v8bf*)(k + (H0 + i) * HDIM + c * 8);
#pragma unroll
      for (int t = 0; t < 8; ++t) s[i] += qf[t] * (float)kc[t];
    }
  }
  const float scale = 0.08838834764831845f;  // 1/sqrt(128)
#pragma unroll
  for (int i = 0; i < 8; ++i) sc[wave][h][H0 + i] = s[i] * scale;
  __syncthreads();

  if (lane < NHEADS) {
    float mx = -1e30f;
#pragma unroll
    for (int j = 0; j < NHEADS; ++j) mx = fmaxf(mx, sc[wave][lane][j]);
    float e[NHEADS];
    float sum = 0.f;
#pragma unroll
    for (int j = 0; j < NHEADS; ++j) {
      e[j] = __expf(sc[wave][lane][j] - mx);
      sum += e[j];
    }
    float inv = 1.0f / sum;
#pragma unroll
    for (int j = 0; j < NHEADS; ++j) sc[wave][lane][j] = e[j] * inv;
  }
  __syncthreads();

  const int d0 = (lane & 1) * 64;
  float w[NHEADS];
#pragma unroll
  for (int H = 0; H < NHEADS; ++H) w[H] = sc[wave][h][H];
  for (int c = 0; c < 8; ++c) {      // lane's 64 outputs in 8-wide chunks
    int d = d0 + c * 8;
    float o[8] = {0.f, 0.f, 0.f, 0.f, 0.f, 0.f, 0.f, 0.f};
#pragma unroll
    for (int H = 0; H < NHEADS; ++H) {
      v8bf vc = *(const v8bf*)(v + H * HDIM + d);
#pragma unroll
      for (int t = 0; t < 8; ++t) o[t] += w[H] * (float)vc[t];
    }
    v8bf ov;
#pragma unroll
    for (int t = 0; t < 8; ++t) ov[t] = (__bf16)o[t];
    *(v8bf*)(attn + (size_t)pos * EMBED + h * HDIM + d) = ov;
  }
}

// ---------------------------------------------------------------------------
extern "C" void kernel_launch(void* const* d_in, const int* in_sizes, int n_in,
                              void* d_out, int out_size, void* d_ws,
                              size_t ws_size, hipStream_t stream) {
  (void)in_sizes; (void)n_in; (void)out_size; (void)ws_size;
  const float* x     = (const float*)d_in[0];
  const float* W_qkv = (const float*)d_in[1];
  const float* W_out = (const float*)d_in[2];
  const float* b_out = (const float*)d_in[3];

  // workspace layout (bf16): x, W_qkv, W_out, qkv, attn  (~352 MB total)
  char* ws = (char*)d_ws;
  __bf16* xb    = (__bf16*)ws;  ws += (size_t)MROWS * EMBED * 2;
  __bf16* wqkvb = (__bf16*)ws;  ws += (size_t)EMBED * NQKV * 2;
  __bf16* woutb = (__bf16*)ws;  ws += (size_t)EMBED * EMBED * 2;
  __bf16* qkvb  = (__bf16*)ws;  ws += (size_t)MROWS * NQKV * 2;
  __bf16* attnb = (__bf16*)ws;

  const int nx = MROWS * EMBED;
  const int nq = EMBED * NQKV;
  const int nw = EMBED * EMBED;
  f32_to_bf16_kernel<<<(nx + 255) / 256, 256, 0, stream>>>(x, xb, nx);
  f32_to_bf16_kernel<<<(nq + 255) / 256, 256, 0, stream>>>(W_qkv, wqkvb, nq);
  f32_to_bf16_kernel<<<(nw + 255) / 256, 256, 0, stream>>>(W_out, woutb, nw);

  dim3 g1(NQKV / BN, MROWS / BM);   // 48 x 128
  gemm_bf16_wmma<true><<<g1, 256, 0, stream>>>(xb, wqkvb, (void*)qkvb, nullptr,
                                               MROWS, NQKV, EMBED);

  attn_softmax_kernel<<<MROWS / 4, 128, 0, stream>>>(qkvb, attnb);

  dim3 g2(EMBED / BN, MROWS / BM);  // 16 x 128
  gemm_bf16_wmma<false><<<g2, 256, 0, stream>>>(attnb, woutb, d_out, b_out,
                                                MROWS, EMBED, EMBED);
}
